// MambaBlock_71983651881629
// MI455X (gfx1250) — matrix operand order repaired and checked by reading, not compiled, not run on hardware
//
#include <hip/hip_runtime.h>
#include <hip/hip_bf16.h>

// ---------------- problem constants ----------------
#define DMODEL 1024
#define DSTATE 16
#define DCONV  4
#define DI     2048          // D_INNER
#define DTRANK 64
#define BATCH  2
#define SEQ    2048
#define MROWS  (BATCH * SEQ) // 4096

// ---------------- WMMA types ----------------
typedef __attribute__((ext_vector_type(16))) __bf16 v16bf;
typedef __attribute__((ext_vector_type(8)))  float  v8f;

union FragBF { v16bf v; uint4 q[2]; };

static __device__ __forceinline__ unsigned short f32_to_bf16(float f) {
    unsigned int u = __float_as_uint(f);
    unsigned int r = 0x7FFFu + ((u >> 16) & 1u);
    return (unsigned short)((u + r) >> 16);
}

// ---------------- async global->LDS staging (CDNA5 ASYNCcnt path) ----------------
#if defined(__gfx1250__) && __has_builtin(__builtin_amdgcn_global_load_async_to_lds_b128)
#define HAVE_ASYNC_LDS 1
#else
#define HAVE_ASYNC_LDS 0
#endif

typedef int async_v4i __attribute__((ext_vector_type(4)));
typedef __attribute__((address_space(1))) async_v4i* gptr_v4i;
typedef __attribute__((address_space(3))) async_v4i* lptr_v4i;

static __device__ __forceinline__ void stage16(unsigned short* lds, const unsigned short* g) {
#if HAVE_ASYNC_LDS
    __builtin_amdgcn_global_load_async_to_lds_b128((gptr_v4i)g, (lptr_v4i)lds, 0, 0);
#else
    *(uint4*)lds = *(const uint4*)g;
#endif
}

static __device__ __forceinline__ void wait_stage() {
#if HAVE_ASYNC_LDS
#if __has_builtin(__builtin_amdgcn_s_wait_asynccnt)
    __builtin_amdgcn_s_wait_asynccnt(0);
#else
    asm volatile("s_wait_asynccnt 0x0" ::: "memory");
#endif
#endif
}

// ---------------- generic fp32 -> bf16 convert ----------------
__global__ __launch_bounds__(256) void cvt_bf16_kernel(const float* __restrict__ in,
                                                       unsigned short* __restrict__ out,
                                                       int n) {
    int i = blockIdx.x * 256 + threadIdx.x;
    if (i < n) out[i] = f32_to_bf16(in[i]);
}

// ---------------- LayerNorm -> bf16 ----------------
__global__ __launch_bounds__(256) void ln_bf16_kernel(const float* __restrict__ x,
                                                      const float* __restrict__ w,
                                                      const float* __restrict__ b,
                                                      unsigned short* __restrict__ out) {
    __shared__ float red[256];
    const int row = blockIdx.x;
    const float* xr = x + (size_t)row * DMODEL;
    float s = 0.f, ss = 0.f;
    for (int i = threadIdx.x; i < DMODEL; i += 256) {
        float v = xr[i]; s += v; ss += v * v;
    }
    red[threadIdx.x] = s; __syncthreads();
    for (int o = 128; o > 0; o >>= 1) {
        if (threadIdx.x < o) red[threadIdx.x] += red[threadIdx.x + o];
        __syncthreads();
    }
    const float mean = red[0] * (1.0f / DMODEL);
    __syncthreads();
    red[threadIdx.x] = ss; __syncthreads();
    for (int o = 128; o > 0; o >>= 1) {
        if (threadIdx.x < o) red[threadIdx.x] += red[threadIdx.x + o];
        __syncthreads();
    }
    const float var = red[0] * (1.0f / DMODEL) - mean * mean;
    const float inv = rsqrtf(var + 1e-5f);
    for (int i = threadIdx.x; i < DMODEL; i += 256) {
        float v = (xr[i] - mean) * inv * w[i] + b[i];
        out[(size_t)row * DMODEL + i] = f32_to_bf16(v);
    }
}

// ---------------- WMMA bf16 GEMM (double-buffered async LDS staging) ----------------
// C[M][N] = A[M][K] * Bw[N][K]^T   (A: bf16 row-major stride lda, Bw: bf16 row-major stride K)
// MODE 0: plain f32 store
// MODE 1: v = softplus(v + bias[n])        (dt path)
// MODE 2: v += res[m*ldc + n]              (final residual)
// MODE 3: f32 store + bf16 copy to Cb      (x_dbl path)
template <int MODE>
__global__ __launch_bounds__(256) void gemm_bf16_wmma(
    const unsigned short* __restrict__ A, int lda,
    const unsigned short* __restrict__ Bw,
    float* __restrict__ C, int ldc,
    unsigned short* __restrict__ Cb,
    const float* __restrict__ bias,
    const float* __restrict__ res,
    int M, int N, int K) {
    (void)M;
    constexpr int BM = 128, BN = 128, BK = 32;
    __shared__ __align__(16) unsigned short sA[2][BM * BK]; // 2 x 8 KB
    __shared__ __align__(16) unsigned short sB[2][BN * BK]; // 2 x 8 KB

    const int tid  = threadIdx.x;
    const int lane = tid & 31;
    const int wave = tid >> 5;
    const int wm   = wave & 1;   // wave row  -> rows wm*64 .. +63
    const int wn   = wave >> 1;  // wave col  -> cols wn*32 .. +31
    const int lr   = lane & 15;  // row/col inside 16x16 tile
    const int lh   = lane >> 4;  // half select

    const int mBase = blockIdx.y * BM;
    const int nBase = blockIdx.x * BN;

    // zero-fill B staging (rows with n >= N are never overwritten -> stay 0)
    {
        const uint4 z = make_uint4(0u, 0u, 0u, 0u);
#pragma unroll
        for (int r = 0; r < 2; ++r) {
            int v = tid + r * 256;
            *(uint4*)&sB[0][v * 8] = z;
            *(uint4*)&sB[1][v * 8] = z;
        }
    }
    __syncthreads();

    const int numK = K / BK;

    // issue one K-chunk worth of async loads into buffer `buf`
    auto stage_chunk = [&](int kt, int buf) {
        const int kBase = kt * BK;
#pragma unroll
        for (int r = 0; r < 2; ++r) {
            int v   = tid + r * 256;
            int row = v >> 2;
            int c8  = (v & 3) * 8;
            stage16(&sA[buf][row * BK + c8],
                    &A[(size_t)(mBase + row) * lda + kBase + c8]);
        }
#pragma unroll
        for (int r = 0; r < 2; ++r) {
            int v   = tid + r * 256;
            int row = v >> 2;
            int c8  = (v & 3) * 8;
            int gn  = nBase + row;
            if (gn < N)
                stage16(&sB[buf][row * BK + c8],
                        &Bw[(size_t)gn * K + kBase + c8]);
        }
    };

    v8f acc[4][2];
    const v8f vzero = {0.f, 0.f, 0.f, 0.f, 0.f, 0.f, 0.f, 0.f};
#pragma unroll
    for (int i = 0; i < 4; ++i)
#pragma unroll
        for (int j = 0; j < 2; ++j) acc[i][j] = vzero;

    // preload chunk 0
    stage_chunk(0, 0);
    wait_stage();
    __syncthreads();

    for (int kt = 0; kt < numK; ++kt) {
        const int buf = kt & 1;
        // overlap: start filling the other buffer while we compute this one
        if (kt + 1 < numK) stage_chunk(kt + 1, buf ^ 1);

        // --- assemble fragments per ISA 7.12.2 layouts ---
        FragBF af[4], bfg[2];
#pragma unroll
        for (int i = 0; i < 4; ++i) {
            int row = wm * 64 + i * 16 + lr;          // M = lane%16
            // A (16-bit 16x32): q0 -> K = lh*8..+7 ; q1 -> K = 16+lh*8..+7
            af[i].q[0] = *(const uint4*)&sA[buf][row * BK + lh * 8];
            af[i].q[1] = *(const uint4*)&sA[buf][row * BK + 16 + lh * 8];
        }
#pragma unroll
        for (int j = 0; j < 2; ++j) {
            int col = wn * 32 + j * 16 + lr;          // N = lane%16
            // B (16-bit 32x16): q0 -> K = lh*8..+7 ; q1 -> K = 16+lh*8..+7
            bfg[j].q[0] = *(const uint4*)&sB[buf][col * BK + lh * 8];
            bfg[j].q[1] = *(const uint4*)&sB[buf][col * BK + 16 + lh * 8];
        }
#pragma unroll
        for (int i = 0; i < 4; ++i)
#pragma unroll
            for (int j = 0; j < 2; ++j)
                acc[i][j] = __builtin_amdgcn_wmma_f32_16x16x32_bf16(
                    false, af[i].v, false, bfg[j].v, (short)0, acc[i][j], false, false);

        // next buffer must be complete and everyone done reading current one
        if (kt + 1 < numK) wait_stage();
        __syncthreads();
    }

    // --- epilogue: D layout lane -> (N = lane%16, M = v + 8*(lane/16)) ---
#pragma unroll
    for (int i = 0; i < 4; ++i) {
#pragma unroll
        for (int j = 0; j < 2; ++j) {
            const int n = nBase + wn * 32 + j * 16 + lr;
            if (n < N) {
#pragma unroll
                for (int e = 0; e < 8; ++e) {
                    const int m = mBase + wm * 64 + i * 16 + e + 8 * lh;
                    float v = acc[i][j][e];
                    if (MODE == 1) {
                        v += bias[n];
                        v = (v > 20.f) ? v : __logf(1.f + __expf(v));
                    }
                    if (MODE == 2) v += res[(size_t)m * ldc + n];
                    C[(size_t)m * ldc + n] = v;
                    if (MODE == 3) Cb[(size_t)m * ldc + n] = f32_to_bf16(v);
                }
            }
        }
    }
}

// ---------------- depthwise causal conv(4) + SiLU ----------------
__global__ __launch_bounds__(256) void conv_silu_kernel(const float* __restrict__ xz,
                                                        const float* __restrict__ cw,
                                                        const float* __restrict__ cb,
                                                        float* __restrict__ uf,
                                                        unsigned short* __restrict__ ub) {
    int idx = blockIdx.x * 256 + threadIdx.x;
    if (idx >= MROWS * DI) return;
    const int d  = idx % DI;
    const int bl = idx / DI;
    const int l  = bl % SEQ;
    const int bb = bl / SEQ;
    float acc = cb[d];
#pragma unroll
    for (int j = 0; j < DCONV; ++j) {
        int ls = l - (DCONV - 1) + j;
        if (ls >= 0)
            acc += cw[d * DCONV + j] * xz[(size_t)(bb * SEQ + ls) * (2 * DI) + d];
    }
    float u = acc / (1.f + __expf(-acc)); // silu
    uf[(size_t)bl * DI + d] = u;
    ub[(size_t)bl * DI + d] = f32_to_bf16(u);
}

// ---------------- selective scan (recurrent over SEQ) ----------------
__global__ __launch_bounds__(256) void scan_kernel(const float* __restrict__ dt_sp,
                                                   const float* __restrict__ xdbl,
                                                   const float* __restrict__ uf,
                                                   const float* __restrict__ xz,
                                                   const float* __restrict__ A_log,
                                                   const float* __restrict__ Dv,
                                                   unsigned short* __restrict__ yb) {
    int gid = blockIdx.x * 256 + threadIdx.x;
    if (gid >= BATCH * DI) return;
    const int bb = gid / DI;
    const int d  = gid % DI;
    float a[DSTATE], h[DSTATE];
#pragma unroll
    for (int s = 0; s < DSTATE; ++s) {
        a[s] = -__expf(A_log[d * DSTATE + s]);
        h[s] = 0.f;
    }
    const float Dd = Dv[d];
    for (int l = 0; l < SEQ; ++l) {
        const size_t bl = (size_t)bb * SEQ + l;
        const float dtv = dt_sp[bl * DI + d];
        const float uv  = uf[bl * DI + d];
        const float* xd = xdbl + bl * (DTRANK + 2 * DSTATE);
        const float du  = dtv * uv;
        float y = 0.f;
#pragma unroll
        for (int s = 0; s < DSTATE; ++s) {
            h[s] = __expf(dtv * a[s]) * h[s] + du * xd[DTRANK + s];
            y += h[s] * xd[DTRANK + DSTATE + s];
        }
        const float zv  = xz[bl * (2 * DI) + DI + d];
        const float out = (y + Dd * uv) * (zv / (1.f + __expf(-zv)));
        yb[bl * DI + d] = f32_to_bf16(out);
    }
}

// ---------------- host-side launch ----------------
static inline size_t alignup(size_t x) { return (x + 255) & ~(size_t)255; }

extern "C" void kernel_launch(void* const* d_in, const int* in_sizes, int n_in,
                              void* d_out, int out_size, void* d_ws, size_t ws_size,
                              hipStream_t stream) {
    (void)in_sizes; (void)n_in; (void)out_size; (void)ws_size;
    const float* x      = (const float*)d_in[0];
    const float* ln_w   = (const float*)d_in[1];
    const float* ln_b   = (const float*)d_in[2];
    const float* W_in   = (const float*)d_in[3];
    const float* conv_w = (const float*)d_in[4];
    const float* conv_b = (const float*)d_in[5];
    const float* W_x    = (const float*)d_in[6];
    const float* W_dt   = (const float*)d_in[7];
    const float* b_dt   = (const float*)d_in[8];
    const float* A_log  = (const float*)d_in[9];
    const float* Dvec   = (const float*)d_in[10];
    const float* W_out  = (const float*)d_in[11];
    float* out = (float*)d_out;

    // ---- workspace partition ----
    char* p = (char*)d_ws;
    size_t off = 0;
    auto carve = [&](size_t bytes) { void* r = p + off; off += alignup(bytes); return r; };
    unsigned short* xn_bf   = (unsigned short*)carve((size_t)MROWS * DMODEL * 2);
    unsigned short* Win_bf  = (unsigned short*)carve((size_t)2 * DI * DMODEL * 2);
    unsigned short* Wx_bf   = (unsigned short*)carve((size_t)(DTRANK + 2 * DSTATE) * DI * 2);
    unsigned short* Wdt_bf  = (unsigned short*)carve((size_t)DI * DTRANK * 2);
    unsigned short* Wout_bf = (unsigned short*)carve((size_t)DMODEL * DI * 2);
    float*          xz      = (float*)carve((size_t)MROWS * 2 * DI * 4);
    float*          u_f     = (float*)carve((size_t)MROWS * DI * 4);
    unsigned short* u_bf    = (unsigned short*)carve((size_t)MROWS * DI * 2);
    float*          xdbl_f  = (float*)carve((size_t)MROWS * (DTRANK + 2 * DSTATE) * 4);
    unsigned short* xdbl_bf = (unsigned short*)carve((size_t)MROWS * (DTRANK + 2 * DSTATE) * 2);
    float*          dt_sp   = (float*)carve((size_t)MROWS * DI * 4);
    unsigned short* y_bf    = (unsigned short*)carve((size_t)MROWS * DI * 2);

    // ---- 1. weight conversions to bf16 ----
    auto cvt = [&](const float* src, unsigned short* dst, int n) {
        cvt_bf16_kernel<<<(n + 255) / 256, 256, 0, stream>>>(src, dst, n);
    };
    cvt(W_in,  Win_bf,  2 * DI * DMODEL);
    cvt(W_x,   Wx_bf,   (DTRANK + 2 * DSTATE) * DI);
    cvt(W_dt,  Wdt_bf,  DI * DTRANK);
    cvt(W_out, Wout_bf, DMODEL * DI);

    // ---- 2. layernorm -> bf16 activations ----
    ln_bf16_kernel<<<MROWS, 256, 0, stream>>>(x, ln_w, ln_b, xn_bf);

    // ---- 3. xz = xn @ W_in^T  (4096 x 1024 x 4096) ----
    {
        dim3 grid((2 * DI) / 128, MROWS / 128);
        gemm_bf16_wmma<0><<<grid, 256, 0, stream>>>(xn_bf, DMODEL, Win_bf, xz, 2 * DI,
                                                    nullptr, nullptr, nullptr,
                                                    MROWS, 2 * DI, DMODEL);
    }

    // ---- 4. depthwise conv + silu -> u (f32 + bf16) ----
    conv_silu_kernel<<<(MROWS * DI + 255) / 256, 256, 0, stream>>>(xz, conv_w, conv_b, u_f, u_bf);

    // ---- 5. x_dbl = u @ W_x^T  (4096 x 2048 x 96), dual f32+bf16 store ----
    {
        const int N = DTRANK + 2 * DSTATE; // 96
        dim3 grid((N + 127) / 128, MROWS / 128);
        gemm_bf16_wmma<3><<<grid, 256, 0, stream>>>(u_bf, DI, Wx_bf, xdbl_f, N,
                                                    xdbl_bf, nullptr, nullptr,
                                                    MROWS, N, DI);
    }

    // ---- 6. dt = softplus(x_dbl[:, :64] @ W_dt^T + b_dt)  (4096 x 64 x 2048) ----
    {
        dim3 grid(DI / 128, MROWS / 128);
        gemm_bf16_wmma<1><<<grid, 256, 0, stream>>>(xdbl_bf, DTRANK + 2 * DSTATE, Wdt_bf,
                                                    dt_sp, DI, nullptr, b_dt, nullptr,
                                                    MROWS, DI, DTRANK);
    }

    // ---- 7. selective scan + u*D + silu(z) gate -> y_bf ----
    scan_kernel<<<(BATCH * DI + 255) / 256, 256, 0, stream>>>(dt_sp, xdbl_f, u_f, xz,
                                                              A_log, Dvec, y_bf);

    // ---- 8. out = x + y @ W_out^T  (4096 x 2048 x 1024) ----
    {
        dim3 grid(DMODEL / 128, MROWS / 128);
        gemm_bf16_wmma<2><<<grid, 256, 0, stream>>>(y_bf, DI, Wout_bf, out, DMODEL,
                                                    nullptr, nullptr, x,
                                                    MROWS, DMODEL, DI);
    }
}
